// Text2SemanticEncoder_13511967113252
// MI455X (gfx1250) — compile-verified, matrix operand order checked by reading, not run on hardware
//
#include <hip/hip_runtime.h>
#include <hip/hip_bf16.h>

typedef _Float16 v16h __attribute__((ext_vector_type(16)));
typedef float    v8f  __attribute__((ext_vector_type(8)));
typedef int      v4i_ls __attribute__((vector_size(16)));   // matches builtin's int4 param

constexpr int cB = 4, cXL = 128, cYL = 1024;
constexpr int cD = 512, cH = 8, cHD = 64, cL = 3, cDFF = 2048;
constexpr int cS  = cXL + 1 + cYL;     // 1153
constexpr int cSP = 1184;              // S padded to multiple of 32
constexpr int cBSP = cB * cSP;         // 4736 (multiple of 128)
constexpr int cOV = 1025, cOVP = 1040, cOVA = 1088; // logits N, ldc, allocated rows
constexpr int cMASK = 1024, cEOS = 1024;
constexpr float cNEG = -1e9f;

union Frag { v16h v; _Float16 h[16]; uint4 q[2]; };

static __device__ __forceinline__ v8f wmma32(v16h a, v16h b, v8f c) {
  // D = A(16x32 f16) * B(32x16 f16) + C(16x16 f32)
  return __builtin_amdgcn_wmma_f32_16x16x32_f16(false, a, false, b, (short)0, c,
                                                false, false);
}

// gfx1250 async global->LDS copy (ASYNCcnt path); guarded for toolchain drift.
#if defined(__HIP_DEVICE_COMPILE__) &&                                         \
    __has_builtin(__builtin_amdgcn_global_load_async_to_lds_b128) &&           \
    __has_builtin(__builtin_amdgcn_s_wait_asynccnt)
#define GLD_ASYNC(dst, src)                                                    \
  __builtin_amdgcn_global_load_async_to_lds_b128(                              \
      (__attribute__((address_space(1))) v4i_ls*)(src),                        \
      (__attribute__((address_space(3))) v4i_ls*)(dst), 0, 0)
#define ASYNC_WAIT() __builtin_amdgcn_s_wait_asynccnt(0)
#else
#define GLD_ASYNC(dst, src) (*(uint4*)(dst) = *(const uint4*)(src))
#define ASYNC_WAIT() ((void)0)
#endif

// ---------------------------------------------------------------- utilities
__global__ void k_zero_f32(float* __restrict__ d, long n) {
  for (long i = blockIdx.x * (long)blockDim.x + threadIdx.x; i < n;
       i += (long)gridDim.x * blockDim.x) d[i] = 0.f;
}
__global__ void k_zero_f16(_Float16* __restrict__ d, long n) {
  for (long i = blockIdx.x * (long)blockDim.x + threadIdx.x; i < n;
       i += (long)gridDim.x * blockDim.x) d[i] = (_Float16)0.f;
}
__global__ void k_f32_to_f16(const float* __restrict__ s, _Float16* __restrict__ d, long n) {
  for (long i = blockIdx.x * (long)blockDim.x + threadIdx.x; i < n;
       i += (long)gridDim.x * blockDim.x) d[i] = (_Float16)s[i];
}

static __device__ __forceinline__ float pe_val(int pos, int d) {
  // sinusoid: even d -> sin(pos*exp(d*-ln(1e4)/D)), odd d -> cos(... (d-1) ...)
  float f = __expf((float)(d & ~1) * (-9.2103403719761836f / (float)cD));
  float a = (float)pos * f;
  return (d & 1) ? __cosf(a) : __sinf(a);
}

// ---------------------------------------------------------------- embedding
__global__ void k_embed(const int* __restrict__ x, const int* __restrict__ y,
                        const unsigned char* __restrict__ ymask,
                        const float* __restrict__ emo_feat,
                        const float* __restrict__ text_emb,
                        const float* __restrict__ audio_emb,
                        const float* __restrict__ type_emb,
                        const float* __restrict__ emo_W, const float* __restrict__ emo_b,
                        const float* __restrict__ alpha_text,
                        const float* __restrict__ alpha_audio,
                        float* __restrict__ hf32, _Float16* __restrict__ hf16) {
  long n = (long)cBSP * cD;
  for (long idx = blockIdx.x * (long)blockDim.x + threadIdx.x; idx < n;
       idx += (long)gridDim.x * blockDim.x) {
    int d = (int)(idx % cD);
    long r = idx / cD;
    int s = (int)(r % cSP);
    int b = (int)(r / cSP);
    float v;
    if (s >= cS) {
      v = 0.f;                                   // padded rows stay zero
    } else if (s < cXL) {
      v = text_emb[(size_t)x[b * cXL + s] * cD + d] + type_emb[d]
          + alpha_text[0] * pe_val(s, d);
    } else if (s == cXL) {
      float acc = emo_b[d];
      const float* w = emo_W + (size_t)d * 768;
      const float* f = emo_feat + (size_t)b * 768;
      for (int k = 0; k < 768; k++) acc += w[k] * f[k];
      v = acc + type_emb[cD + d];
    } else {
      int yp = s - cXL - 1;
      int tok = ymask[b * cYL + yp] ? cMASK : y[b * cYL + yp];
      v = audio_emb[(size_t)tok * cD + d] + type_emb[2 * cD + d]
          + alpha_audio[0] * pe_val(yp, d);
    }
    hf32[idx] = v;
    hf16[idx] = (_Float16)v;
  }
}

// ---------------------------------------------------------------- attn bias
__global__ void k_bias(const int* __restrict__ x1l, const int* __restrict__ x2l,
                       const int* __restrict__ y1l, const int* __restrict__ y2l,
                       const unsigned char* __restrict__ ymask,
                       float* __restrict__ bias) {
  long n = (long)cB * cSP * cSP;
  for (long idx = blockIdx.x * (long)blockDim.x + threadIdx.x; idx < n;
       idx += (long)gridDim.x * blockDim.x) {
    int c = (int)(idx % cSP);
    long r2 = idx / cSP;
    int r = (int)(r2 % cSP);
    int b = (int)(r2 / cSP);
    int x1 = x1l[b], xlen = x1 + 1 + x2l[b];
    int y1 = y1l[b], ylen = y1 + 1 + y2l[b];
    bool allowed = false;
    if (c < cS && r < cS) {
      if (r < cXL) {
        allowed = (r < x1) ? (c < x1) : (c >= x1 && c < xlen);
      } else if (r == cXL) {
        allowed = (c == cXL);
      } else {
        int ry = r - cXL - 1;
        if (ry < y1) {
          allowed = (c < x1) || (c > cXL && c <= cXL + y1);
        } else if (ry == y1) {
          allowed = (c < xlen) || (c == cXL) || (c > cXL && c <= cXL + y1 + 1);
        } else if (ry < ylen) {
          allowed = (c < xlen) || (c == cXL) || (c > cXL && c <= cXL + ylen);
          if (allowed && c > cXL && ymask[b * cYL + (c - cXL - 1)]) allowed = false;
        }
      }
    }
    bias[idx] = allowed ? 0.f : cNEG;
  }
}

// ---------------------------------------------------------------- WMMA GEMM
// C[M,N](f32,ldc) [+f16 copy] = A[M,K](f16,lda) @ Bt[N,K](f16,ldb)^T + bias[N], opt ReLU.
// Block = 128 threads (4 waves). Block tile 128(M)x64(N); wave tile 32(M)x64(N).
// A/B K-slices staged in LDS via async global->LDS copies, double-buffered:
// copies for step i+1 are in flight while WMMAs consume step i from LDS.
__global__ void __launch_bounds__(128)
k_gemm(const _Float16* __restrict__ A, int lda,
       const _Float16* __restrict__ Bt, int ldb,
       float* __restrict__ C, _Float16* __restrict__ Ch, int ldc,
       const float* __restrict__ bias, int relu,
       int M, int N, int K) {
  __shared__ _Float16 Abuf[2][128 * 32];
  __shared__ _Float16 Bbuf[2][64 * 32];
  int tid = threadIdx.x;
  int lane = tid & 31;
  int wave = tid >> 5;
  int col = lane & 15;
  int half = lane >> 4;
  int mblk = blockIdx.y * 128;
  int n0 = blockIdx.x * 64;

  v8f acc[2][4];
#pragma unroll
  for (int mt = 0; mt < 2; mt++)
#pragma unroll
    for (int t = 0; t < 4; t++)
#pragma unroll
      for (int j = 0; j < 8; j++) acc[mt][t][j] = 0.f;

  auto issue = [&](int buf, int kk) {
    // A slice: 128 rows x 32 halves = 512 16B-chunks, 4 per thread
#pragma unroll
    for (int i = 0; i < 4; i++) {
      int id = tid + i * 128;
      int row = id >> 2, q = id & 3;
      GLD_ASYNC(&Abuf[buf][row * 32 + q * 8],
                A + (size_t)(mblk + row) * lda + kk + q * 8);
    }
    // B slice: 64 rows x 32 halves = 256 16B-chunks, 2 per thread
#pragma unroll
    for (int i = 0; i < 2; i++) {
      int id = tid + i * 128;
      int row = id >> 2, q = id & 3;
      GLD_ASYNC(&Bbuf[buf][row * 32 + q * 8],
                Bt + (size_t)(n0 + row) * ldb + kk + q * 8);
    }
  };

  int nk = K >> 5;
  issue(0, 0);
  for (int it = 0; it < nk; ++it) {
    ASYNC_WAIT();
    __syncthreads();
    int cur = it & 1;
    if (it + 1 < nk) issue(cur ^ 1, (it + 1) << 5);

    Frag a0, a1;
    const _Float16* ab = &Abuf[cur][(wave * 32 + col) * 32 + half * 8];
    a0.q[0] = *(const uint4*)(ab);
    a0.q[1] = *(const uint4*)(ab + 16);
    a1.q[0] = *(const uint4*)(ab + 512);        // +16 rows * 32 halves
    a1.q[1] = *(const uint4*)(ab + 512 + 16);
#pragma unroll
    for (int t = 0; t < 4; t++) {
      Frag bf;
      const _Float16* bb = &Bbuf[cur][(t * 16 + col) * 32 + half * 8];
      bf.q[0] = *(const uint4*)(bb);
      bf.q[1] = *(const uint4*)(bb + 16);
      acc[0][t] = wmma32(a0.v, bf.v, acc[0][t]);
      acc[1][t] = wmma32(a1.v, bf.v, acc[1][t]);
    }
  }

#pragma unroll
  for (int mt = 0; mt < 2; mt++) {
#pragma unroll
    for (int t = 0; t < 4; t++) {
      int nn = n0 + t * 16 + col;
      if (nn >= N) continue;
      float bv = bias ? bias[nn] : 0.f;
#pragma unroll
      for (int j = 0; j < 8; j++) {
        int row = mblk + wave * 32 + mt * 16 + half * 8 + j;
        if (row < M) {
          float v = acc[mt][t][j] + bv;
          if (relu) v = fmaxf(v, 0.f);
          if (C)  C[(size_t)row * ldc + nn] = v;
          if (Ch) Ch[(size_t)row * ldc + nn] = (_Float16)v;
        }
      }
    }
  }
}

// ------------------------------------------------------- split/convert QKV
__global__ void k_prep(const float* __restrict__ qkv, _Float16* __restrict__ qh,
                       _Float16* __restrict__ kh, _Float16* __restrict__ vTh) {
  long n = (long)cBSP * cD;
  for (long idx = blockIdx.x * (long)blockDim.x + threadIdx.x; idx < n;
       idx += (long)gridDim.x * blockDim.x) {
    int d = (int)(idx % cD);
    long r = idx / cD;
    int s = (int)(r % cSP);
    int b = (int)(r / cSP);
    int hh = d >> 6, dd = d & 63;
    size_t base = (size_t)r * (3 * cD);
    float q = qkv[base + d];
    float k = qkv[base + cD + d];
    float v = qkv[base + 2 * cD + d];
    size_t bh = (size_t)(b * cH + hh);
    qh[(bh * cSP + s) * cHD + dd] = (_Float16)(q * 0.125f);  // 1/sqrt(HD) folded in
    kh[(bh * cSP + s) * cHD + dd] = (_Float16)k;
    vTh[(bh * cHD + dd) * cSP + s] = (_Float16)v;            // V transposed
  }
}

// --------------------------------------------------- flash attention (P^T)
// One wave per (b,h, 16-row Q tile). Scores computed transposed (K·Qᵀ) so the
// softmax row index == lane%16, and two Pᵀ C-tiles == one WMMA B-fragment.
__global__ void k_attn(const _Float16* __restrict__ qh, const _Float16* __restrict__ kh,
                       const _Float16* __restrict__ vTh, const float* __restrict__ bias,
                       _Float16* __restrict__ attnh) {
  int qt = blockIdx.x;
  int bh = blockIdx.y;
  int b = bh / cH, h = bh % cH;
  int lane = threadIdx.x & 31;
  int col = lane & 15;     // q index within tile / N index
  int half = lane >> 4;
  int q0 = qt * 16;
  const _Float16* qbase = qh + (size_t)bh * cSP * cHD;
  const _Float16* kbase = kh + (size_t)bh * cSP * cHD;
  const _Float16* vbase = vTh + (size_t)bh * cHD * cSP;
  const float* bbase = bias + ((size_t)b * cSP + q0 + col) * cSP;

  // Q B-fragments (K = HD = 64 -> two 32-chunks), resident for whole loop
  Frag bq0, bq1;
  {
    const _Float16* p = qbase + (size_t)(q0 + col) * cHD + half * 8;
    bq0.q[0] = *(const uint4*)(p);       bq0.q[1] = *(const uint4*)(p + 16);
    bq1.q[0] = *(const uint4*)(p + 32);  bq1.q[1] = *(const uint4*)(p + 48);
  }
  v8f oacc[4];
#pragma unroll
  for (int t = 0; t < 4; t++)
#pragma unroll
    for (int j = 0; j < 8; j++) oacc[t][j] = 0.f;
  float m = -3.0e38f, l = 0.f;

  for (int kk = 0; kk < cSP; kk += 32) {
    v8f p0, p1;
#pragma unroll
    for (int j = 0; j < 8; j++) { p0[j] = 0.f; p1[j] = 0.f; }
    {  // P^T tiles: A = K rows, B = Q
      const _Float16* pk = kbase + (size_t)(kk + col) * cHD + half * 8;
      Frag a0, a1;
      a0.q[0] = *(const uint4*)(pk);      a0.q[1] = *(const uint4*)(pk + 16);
      a1.q[0] = *(const uint4*)(pk + 32); a1.q[1] = *(const uint4*)(pk + 48);
      p0 = wmma32(a0.v, bq0.v, p0);
      p0 = wmma32(a1.v, bq1.v, p0);
      const _Float16* pk2 = pk + (size_t)16 * cHD;
      a0.q[0] = *(const uint4*)(pk2);      a0.q[1] = *(const uint4*)(pk2 + 16);
      a1.q[0] = *(const uint4*)(pk2 + 32); a1.q[1] = *(const uint4*)(pk2 + 48);
      p1 = wmma32(a0.v, bq0.v, p1);
      p1 = wmma32(a1.v, bq1.v, p1);
    }
    // add bias: element j of p0 <-> key kk+half*8+j ; p1 <-> +16
    const float* bp = bbase + kk + half * 8;
    float4 b0a = *(const float4*)(bp);
    float4 b0b = *(const float4*)(bp + 4);
    float4 b1a = *(const float4*)(bp + 16);
    float4 b1b = *(const float4*)(bp + 20);
    p0[0] += b0a.x; p0[1] += b0a.y; p0[2] += b0a.z; p0[3] += b0a.w;
    p0[4] += b0b.x; p0[5] += b0b.y; p0[6] += b0b.z; p0[7] += b0b.w;
    p1[0] += b1a.x; p1[1] += b1a.y; p1[2] += b1a.z; p1[3] += b1a.w;
    p1[4] += b1b.x; p1[5] += b1b.y; p1[6] += b1b.z; p1[7] += b1b.w;
    // online softmax over this 32-key chunk (per-q stats are per-lane scalars)
    float tmax = p0[0];
#pragma unroll
    for (int j = 1; j < 8; j++) tmax = fmaxf(tmax, p0[j]);
#pragma unroll
    for (int j = 0; j < 8; j++) tmax = fmaxf(tmax, p1[j]);
    tmax = fmaxf(tmax, __shfl_xor(tmax, 16, 32));
    float mnew = fmaxf(m, tmax);
    float corr = __expf(m - mnew);
    m = mnew;
    float ts = 0.f;
#pragma unroll
    for (int j = 0; j < 8; j++) {
      p0[j] = __expf(p0[j] - mnew);
      p1[j] = __expf(p1[j] - mnew);
      ts += p0[j] + p1[j];
    }
    ts += __shfl_xor(ts, 16, 32);
    l = l * corr + ts;
    // probabilities in C-layout are exactly the WMMA B-fragment layout
    Frag pb;
#pragma unroll
    for (int j = 0; j < 8; j++) {
      pb.h[j] = (_Float16)p0[j];
      pb.h[8 + j] = (_Float16)p1[j];
    }
    // O^T += V^T · P^T   (rescale accumulators first)
#pragma unroll
    for (int t = 0; t < 4; t++) {
#pragma unroll
      for (int j = 0; j < 8; j++) oacc[t][j] *= corr;
      const _Float16* pv = vbase + (size_t)(t * 16 + col) * cSP + kk + half * 8;
      Frag av;
      av.q[0] = *(const uint4*)(pv);
      av.q[1] = *(const uint4*)(pv + 16);
      oacc[t] = wmma32(av.v, pb.v, oacc[t]);
    }
  }
  float inv = 1.0f / l;
  _Float16* ob = attnh + (size_t)(b * cSP + q0 + col) * cD + h * cHD + half * 8;
#pragma unroll
  for (int t = 0; t < 4; t++)
#pragma unroll
    for (int j = 0; j < 8; j++)
      ob[t * 16 + j] = (_Float16)(oacc[t][j] * inv);
}

// ---------------------------------------------------- residual + layernorm
__global__ void k_add_ln(float* __restrict__ hbuf, const float* __restrict__ delta,
                         const float* __restrict__ g, const float* __restrict__ bt,
                         _Float16* __restrict__ hh) {
  int row = blockIdx.x;
  int t = threadIdx.x;  // 128 threads * 4 elems
  __shared__ float s1[128], s2[128];
  float x[4];
  float sum = 0.f, sq = 0.f;
  size_t base = (size_t)row * cD + t * 4;
#pragma unroll
  for (int i = 0; i < 4; i++) {
    x[i] = hbuf[base + i] + delta[base + i];
    sum += x[i];
    sq += x[i] * x[i];
  }
  s1[t] = sum; s2[t] = sq;
  __syncthreads();
  for (int o = 64; o > 0; o >>= 1) {
    if (t < o) { s1[t] += s1[t + o]; s2[t] += s2[t + o]; }
    __syncthreads();
  }
  float mu = s1[0] * (1.0f / cD);
  float var = s2[0] * (1.0f / cD) - mu * mu;
  float rs = rsqrtf(var + 1e-5f);
#pragma unroll
  for (int i = 0; i < 4; i++) {
    int c = t * 4 + i;
    float v = (x[i] - mu) * rs * g[c] + bt[c];
    hbuf[base + i] = v;
    hh[base + i] = (_Float16)v;
  }
}

// ------------------------------------------------ CE + top-3 over masked
__global__ void k_ce(const float* __restrict__ logits, const int* __restrict__ y,
                     const unsigned char* __restrict__ ymask, float* __restrict__ accum) {
  int r = blockIdx.x;  // b*YL + yl
  if (!ymask[r]) return;
  const float* row = logits + (size_t)r * cOVP;
  int tgt = y[r];
  int t = threadIdx.x;  // 128
  __shared__ float smx[128], ssum[128];
  __shared__ int scnt[128];
  float mx = -3.0e38f;
  for (int j = t; j < cOV; j += 128) mx = fmaxf(mx, row[j]);
  smx[t] = mx;
  __syncthreads();
  for (int o = 64; o > 0; o >>= 1) {
    if (t < o) smx[t] = fmaxf(smx[t], smx[t + o]);
    __syncthreads();
  }
  mx = smx[0];
  float tv = row[tgt];
  float sum = 0.f; int cnt = 0;
  for (int j = t; j < cOV; j += 128) {
    float v = row[j];
    sum += __expf(v - mx);
    cnt += (v > tv) ? 1 : 0;
  }
  ssum[t] = sum; scnt[t] = cnt;
  __syncthreads();
  for (int o = 64; o > 0; o >>= 1) {
    if (t < o) { ssum[t] += ssum[t + o]; scnt[t] += scnt[t + o]; }
    __syncthreads();
  }
  if (t == 0) {
    float ce = mx + __logf(ssum[0]) - tv;       // lse - logit[target]
    atomicAdd(&accum[0], ce);
    atomicAdd(&accum[1], 1.0f);
    float inc = (tgt != cEOS) ? 1.f : 0.f;
    if (scnt[0] < 3) atomicAdd(&accum[2], inc); // target in top-3
    atomicAdd(&accum[3], inc);
  }
}

// ------------------------------------------------------- duration MLP
__global__ void k_dur(const float* __restrict__ hbuf, const int* __restrict__ y1l,
                      const float* __restrict__ dW1, const float* __restrict__ db1,
                      const float* __restrict__ dW2, const float* __restrict__ db2,
                      const float* __restrict__ dW3, const float* __restrict__ db3,
                      float* __restrict__ accum) {
  int b = blockIdx.x;
  int t = threadIdx.x;  // 256
  __shared__ float cls[cD];
  __shared__ float h1[256];
  __shared__ float h2[128];
  int row = b * cSP + cXL + 1 + y1l[b];
  for (int i = t; i < cD; i += 256) cls[i] = hbuf[(size_t)row * cD + i];
  __syncthreads();
  {
    float s = db1[t];
    const float* w = dW1 + (size_t)t * cD;
    for (int k = 0; k < cD; k++) s += w[k] * cls[k];
    h1[t] = fmaxf(s, 0.f);
  }
  __syncthreads();
  if (t < 128) {
    float s = db2[t];
    const float* w = dW2 + (size_t)t * 256;
    for (int k = 0; k < 256; k++) s += w[k] * h1[k];
    h2[t] = fmaxf(s, 0.f);
  }
  __syncthreads();
  if (t == 0) {
    float s = db3[0];
    for (int k = 0; k < 128; k++) s += dW3[k] * h2[k];
    accum[4 + b] = s;
  }
}

__global__ void k_final(const float* __restrict__ accum, const int* __restrict__ x2l,
                        const int* __restrict__ y2l, float* __restrict__ out) {
  if (threadIdx.x == 0 && blockIdx.x == 0) {
    float token = accum[0] / accum[1];
    float dl = 0.f;
    for (int b = 0; b < cB; b++) {
      float tgt = (float)y2l[b] / (float)x2l[b];
      float e = accum[4 + b] - tgt;
      float ae = fabsf(e);
      dl += (ae <= 1.f) ? 0.5f * e * e : ae - 0.5f;
    }
    dl *= 1.0f / cB;
    out[0] = token + dl;                         // DUR_W = 1.0
    out[1] = accum[2] / fmaxf(accum[3], 1.f);
  }
}

// ================================================================= launch
extern "C" void kernel_launch(void* const* d_in, const int* in_sizes, int n_in,
                              void* d_out, int out_size, void* d_ws, size_t ws_size,
                              hipStream_t stream) {
  (void)in_sizes; (void)n_in; (void)out_size; (void)ws_size;
  const int* x        = (const int*)d_in[0];
  const int* x1_lens  = (const int*)d_in[1];
  const int* x2_lens  = (const int*)d_in[2];
  const int* y        = (const int*)d_in[3];
  const int* y1_lens  = (const int*)d_in[4];
  const int* y2_lens  = (const int*)d_in[5];
  const unsigned char* ymask = (const unsigned char*)d_in[6];  // jax bool -> 1 byte
  const float* emo_feature = (const float*)d_in[7];
  const float* text_emb  = (const float*)d_in[8];
  const float* audio_emb = (const float*)d_in[9];
  const float* type_emb  = (const float*)d_in[10];
  const float* emo_W = (const float*)d_in[11];
  const float* emo_b = (const float*)d_in[12];
  const float* alpha_text  = (const float*)d_in[13];
  const float* alpha_audio = (const float*)d_in[14];
  const float* Wqkv = (const float*)d_in[15];
  const float* bqkv = (const float*)d_in[16];
  const float* Wo   = (const float*)d_in[17];
  const float* bo   = (const float*)d_in[18];
  const float* ln1_g = (const float*)d_in[19];
  const float* ln1_b = (const float*)d_in[20];
  const float* ln2_g = (const float*)d_in[21];
  const float* ln2_b = (const float*)d_in[22];
  const float* W1 = (const float*)d_in[23];
  const float* b1 = (const float*)d_in[24];
  const float* W2 = (const float*)d_in[25];
  const float* b2 = (const float*)d_in[26];
  const float* predict_W = (const float*)d_in[27];
  const float* dW1 = (const float*)d_in[28];
  const float* db1 = (const float*)d_in[29];
  const float* dW2 = (const float*)d_in[30];
  const float* db2 = (const float*)d_in[31];
  const float* dW3 = (const float*)d_in[32];
  const float* db3 = (const float*)d_in[33];

  char* p = (char*)d_ws;
  auto alloc = [&](size_t bytes) -> char* {
    char* r = p;
    p += (bytes + 255) & ~(size_t)255;
    return r;
  };
  float*     hf32  = (float*)alloc((size_t)cBSP * cD * 4);
  _Float16*  hf16  = (_Float16*)alloc((size_t)cBSP * cD * 2);
  float*     biasb = (float*)alloc((size_t)cB * cSP * cSP * 4);
  float*     qkv   = (float*)alloc((size_t)cBSP * 3 * cD * 4);
  _Float16*  qh    = (_Float16*)alloc((size_t)cB * cH * cSP * cHD * 2);
  _Float16*  kh    = (_Float16*)alloc((size_t)cB * cH * cSP * cHD * 2);
  _Float16*  vTh   = (_Float16*)alloc((size_t)cB * cH * cHD * cSP * 2);
  _Float16*  attnh = (_Float16*)alloc((size_t)cBSP * cD * 2);
  float*     obuf  = (float*)alloc((size_t)cBSP * cD * 4);
  _Float16*  ff1h  = (_Float16*)alloc((size_t)cBSP * cDFF * 2);
  float*     ff2   = (float*)alloc((size_t)cBSP * cD * 4);
  _Float16*  WqkvH = (_Float16*)alloc((size_t)cL * 3 * cD * cD * 2);
  _Float16*  WoH   = (_Float16*)alloc((size_t)cL * cD * cD * 2);
  _Float16*  W1H   = (_Float16*)alloc((size_t)cL * cDFF * cD * 2);
  _Float16*  W2H   = (_Float16*)alloc((size_t)cL * cD * cDFF * 2);
  _Float16*  predH = (_Float16*)alloc((size_t)cOVA * cD * 2);
  float*     logits = (float*)alloc((size_t)cB * cYL * cOVP * 4);
  float*     accum  = (float*)alloc(16 * 4);

  k_zero_f32<<<1, 32, 0, stream>>>(accum, 16);

  auto cvt = [&](const float* s, _Float16* d, long n) {
    long blks = (n + 255) / 256;
    if (blks > 8192) blks = 8192;
    k_f32_to_f16<<<(int)blks, 256, 0, stream>>>(s, d, n);
  };
  cvt(Wqkv, WqkvH, (long)cL * 3 * cD * cD);
  cvt(Wo, WoH, (long)cL * cD * cD);
  cvt(W1, W1H, (long)cL * cDFF * cD);
  cvt(W2, W2H, (long)cL * cD * cDFF);
  cvt(predict_W, predH, (long)cOV * cD);
  k_zero_f16<<<64, 256, 0, stream>>>(predH + (size_t)cOV * cD, (long)(cOVA - cOV) * cD);

  k_embed<<<4096, 256, 0, stream>>>(x, y, ymask, emo_feature, text_emb, audio_emb,
                                    type_emb, emo_W, emo_b, alpha_text, alpha_audio,
                                    hf32, hf16);
  k_bias<<<8192, 256, 0, stream>>>(x1_lens, x2_lens, y1_lens, y2_lens, ymask, biasb);

  for (int l = 0; l < cL; l++) {
    // QKV: [BSP,512] x [1536,512]^T
    k_gemm<<<dim3(3 * cD / 64, cBSP / 128), 128, 0, stream>>>(
        hf16, cD, WqkvH + (size_t)l * 3 * cD * cD, cD,
        qkv, ((_Float16*)nullptr), 3 * cD, bqkv + (size_t)l * 3 * cD, 0,
        cBSP, 3 * cD, cD);
    k_prep<<<4096, 256, 0, stream>>>(qkv, qh, kh, vTh);
    k_attn<<<dim3(cSP / 16, cB * cH), 32, 0, stream>>>(qh, kh, vTh, biasb, attnh);
    // O-proj
    k_gemm<<<dim3(cD / 64, cBSP / 128), 128, 0, stream>>>(
        attnh, cD, WoH + (size_t)l * cD * cD, cD,
        obuf, ((_Float16*)nullptr), cD, bo + (size_t)l * cD, 0, cBSP, cD, cD);
    k_add_ln<<<cBSP, 128, 0, stream>>>(hf32, obuf, ln1_g + (size_t)l * cD,
                                       ln1_b + (size_t)l * cD, hf16);
    // FFN up (+ReLU, f16 out only)
    k_gemm<<<dim3(cDFF / 64, cBSP / 128), 128, 0, stream>>>(
        hf16, cD, W1H + (size_t)l * cDFF * cD, cD,
        ((float*)nullptr), ff1h, cDFF, b1 + (size_t)l * cDFF, 1, cBSP, cDFF, cD);
    // FFN down
    k_gemm<<<dim3(cD / 64, cBSP / 128), 128, 0, stream>>>(
        ff1h, cDFF, W2H + (size_t)l * cD * cDFF, cDFF,
        ff2, ((_Float16*)nullptr), cD, b2 + (size_t)l * cD, 0, cBSP, cD, cDFF);
    k_add_ln<<<cBSP, 128, 0, stream>>>(hf32, ff2, ln2_g + (size_t)l * cD,
                                       ln2_b + (size_t)l * cD, hf16);
  }

  // logits over y region, one GEMM per batch (rows contiguous within batch)
  for (int b = 0; b < cB; b++) {
    k_gemm<<<dim3(cOVA / 64, cYL / 128), 128, 0, stream>>>(
        hf16 + ((size_t)(b * cSP) + cXL + 1) * cD, cD, predH, cD,
        logits + (size_t)b * cYL * cOVP, ((_Float16*)nullptr), cOVP,
        ((const float*)nullptr), 0, cYL, cOV, cD);
  }

  k_ce<<<cB * cYL, 128, 0, stream>>>(logits, y, ymask, accum);
  k_dur<<<cB, 256, 0, stream>>>(hf32, y1_lens, dW1, db1, dW2, db2, dW3, db3, accum);
  k_final<<<1, 1, 0, stream>>>(accum, x2_lens, y2_lens, (float*)d_out);
}
